// DTINet1_1331439862049
// MI455X (gfx1250) — matrix operand order repaired and checked by reading, not compiled, not run on hardware
//
#include <hip/hip_runtime.h>
#include <hip/hip_bf16.h>
#include <math.h>

// ---------------------------------------------------------------------------
// Types for CDNA5 WMMA
// ---------------------------------------------------------------------------
typedef __attribute__((ext_vector_type(16))) __bf16 v16bf;
typedef __attribute__((ext_vector_type(8)))  float  v8f;

#define VOX_D 48
#define VOX_H 48
#define VOX_W 32
#define NVOX  (VOX_D * VOX_H * VOX_W)   // 73728

// LDS row padding: rows of 32 bf16 padded to 40 bf16 (80B) so 16B fragment
// reads stride by 5 bank-quads (coprime with 16) -> minimal bank conflicts.
#define LROW 40

// Dynamic-LDS layout (byte offsets; no static LDS so dynamic region starts @0)
#define ACT_ROWS   256
#define WGT_ROWS   64
#define ACT_OFF_B  0
#define WGT_OFF_B  (ACT_ROWS * LROW * 2)              // 20480
#define LDS_BYTES  (WGT_OFF_B + WGT_ROWS * LROW * 2)  // 25600

// ---------------------------------------------------------------------------
// Repack input: x (48,48,32,91) f32 -> act[vox][96] bf16 with abs(), zero pad.
// Also zero-fills the 256B OOB-halo scratch region (branch-free halo loads).
// ---------------------------------------------------------------------------
__global__ void repack_input_kernel(const float* __restrict__ x,
                                    __bf16* __restrict__ act,
                                    __bf16* __restrict__ zbuf, int n) {
    int i = blockIdx.x * blockDim.x + threadIdx.x;
    if (i < 128) zbuf[i] = (__bf16)0.0f;
    if (i >= n) return;
    int vox = i / 96;
    int c   = i - vox * 96;
    float v = (c < 91) ? fabsf(x[(size_t)vox * 91 + c]) : 0.0f;
    act[i] = (__bf16)v;
}

// ---------------------------------------------------------------------------
// Repack weights: w (cout,cin,3,3,3) f32 -> wT[tap][coutPad][cinK] bf16
// ---------------------------------------------------------------------------
__global__ void repack_weights_kernel(const float* __restrict__ w,
                                      __bf16* __restrict__ wT,
                                      int cout, int cin, int coutPad, int cinK) {
    int i = blockIdx.x * blockDim.x + threadIdx.x;
    int total = 27 * coutPad * cinK;
    if (i >= total) return;
    int t  = i / (coutPad * cinK);
    int r  = i - t * coutPad * cinK;
    int co = r / cinK;
    int ci = r - co * cinK;
    float v = 0.0f;
    if (co < cout && ci < cin)
        v = w[((size_t)co * cin + ci) * 27 + t];
    wT[i] = (__bf16)v;
}

// ---------------------------------------------------------------------------
// Implicit-GEMM conv3d (3x3x3, pad 1) with bf16 WMMA, f32 accumulate.
// Block: 256 threads (8 waves). Tile: 64 cout x 256 vox.
// Wave grid 2(M) x 4(N): each wave -> 32 cout x 64 vox = eight 16x16 tiles.
// Staging uses CDNA5 async global->LDS copies (ASYNCcnt) with a zero-filled
// scratch page for halo lanes so EXEC stays uniform (no divergence).
// ---------------------------------------------------------------------------
__global__ __launch_bounds__(256) void conv3d_bf16_wmma_kernel(
    const __bf16* __restrict__ in, int inStride, int cinK,
    __bf16* __restrict__ out, int outStride, int cout,
    const __bf16* __restrict__ wT, int coutPad,
    const float* __restrict__ bias, const __bf16* __restrict__ zbuf, int relu)
{
    extern __shared__ __align__(16) char smem[];
    const __bf16* sAct = (const __bf16*)(smem + ACT_OFF_B);
    const __bf16* sWgt = (const __bf16*)(smem + WGT_OFF_B);

    const int tid       = threadIdx.x;
    const int voxBase   = blockIdx.x * 256;
    const int coutBlock = blockIdx.y * 64;
    const int lid       = tid & 31;
    const int wid       = tid >> 5;
    const int mBase     = (wid & 1) * 32;   // cout offset within block tile
    const int nBase     = (wid >> 1) * 64;  // vox offset within block tile

    v8f acc[2][4];
#pragma unroll
    for (int i = 0; i < 2; ++i)
#pragma unroll
        for (int j = 0; j < 4; ++j)
#pragma unroll
            for (int v = 0; v < 8; ++v) acc[i][j][v] = 0.0f;

    const int nK = cinK >> 5;  // K-steps of 32 per tap

    for (int t = 0; t < 27; ++t) {
        const int kd = t / 9 - 1;
        const int kh = (t / 3) % 3 - 1;
        const int kw = t % 3 - 1;

        for (int kc = 0; kc < nK; ++kc) {
            const int kBase = kc << 5;
            __syncthreads();  // previous tile fully consumed by all waves

            // ---- stage activations: 256 rows x 64B = 1024 x 16B chunks ----
            // CDNA5 direct global->LDS async copy; OOB lanes read the zero page.
#pragma unroll
            for (int it = 0; it < 4; ++it) {
                int c   = tid + it * 256;
                int row = c >> 2;
                int off = (c & 3) << 3;  // bf16 element offset within row
                int vox = voxBase + row;
                int z   = vox / (VOX_H * VOX_W);
                int rem = vox - z * (VOX_H * VOX_W);
                int y   = rem >> 5;
                int x   = rem & 31;
                int zz = z + kd, yy = y + kh, xx = x + kw;
                bool ok = (unsigned)zz < (unsigned)VOX_D &&
                          (unsigned)yy < (unsigned)VOX_H &&
                          (unsigned)xx < (unsigned)VOX_W;
                size_t svox = ((size_t)zz * VOX_H + yy) * VOX_W + xx;
                const __bf16* src =
                    ok ? &in[svox * inStride + kBase + off] : zbuf;
                unsigned ldsOff = (unsigned)(ACT_OFF_B + (row * LROW + off) * 2);
                asm volatile("global_load_async_to_lds_b128 %0, %1, off"
                             :: "v"(ldsOff), "v"(src) : "memory");
            }

            // ---- stage weights: 64 rows x 64B = 256 x 16B chunks ----
            {
                int row = tid >> 2;
                int off = (tid & 3) << 3;
                const __bf16* src =
                    &wT[((size_t)t * coutPad + coutBlock + row) * cinK + kBase + off];
                unsigned ldsOff = (unsigned)(WGT_OFF_B + (row * LROW + off) * 2);
                asm volatile("global_load_async_to_lds_b128 %0, %1, off"
                             :: "v"(ldsOff), "v"(src) : "memory");
            }

            // drain ASYNCcnt, then block-wide barrier -> LDS tile visible
            asm volatile("s_wait_asynccnt 0x0" ::: "memory");
            __syncthreads();

            // ---- load fragments ----
            // A (weights, 16x32 bf16): lane<16 -> K{0..7,16..23}; lane>=16 -> K{8..15,24..31}
            v16bf afrag[2];
#pragma unroll
            for (int mt = 0; mt < 2; ++mt) {
                const __bf16* p =
                    &sWgt[(mBase + mt * 16 + (lid & 15)) * LROW + ((lid >> 4) << 3)];
                union { v16bf v; uint4 q[2]; } u;
                u.q[0] = *(const uint4*)p;
                u.q[1] = *(const uint4*)(p + 16);
                afrag[mt] = u.v;
            }
            // B (acts, 32x16 bf16): col = lane&15; lane<16 -> K{0..7,16..23}; lane>=16 -> K{8..15,24..31}
            v16bf bfrag[4];
#pragma unroll
            for (int nt = 0; nt < 4; ++nt) {
                const __bf16* p =
                    &sAct[(nBase + nt * 16 + (lid & 15)) * LROW + ((lid >> 4) << 3)];
                union { v16bf v; uint4 q[2]; } u;
                u.q[0] = *(const uint4*)p;
                u.q[1] = *(const uint4*)(p + 16);
                bfrag[nt] = u.v;
            }

            // ---- eight WMMAs per wave per K-step ----
#pragma unroll
            for (int mt = 0; mt < 2; ++mt)
#pragma unroll
                for (int nt = 0; nt < 4; ++nt)
                    acc[mt][nt] = __builtin_amdgcn_wmma_f32_16x16x32_bf16(
                        false, afrag[mt], false, bfrag[nt],
                        (short)0, acc[mt][nt], false, false);
        }
    }

    // ---- epilogue: bias + optional ReLU, store bf16 channels-last ----
#pragma unroll
    for (int mt = 0; mt < 2; ++mt) {
        int coutLocal = coutBlock + mBase + mt * 16 + ((lid >> 4) << 3);
#pragma unroll
        for (int nt = 0; nt < 4; ++nt) {
            int vox = voxBase + nBase + nt * 16 + (lid & 15);
            union { __bf16 h[8]; uint4 q; } o;
#pragma unroll
            for (int v = 0; v < 8; ++v) {
                int cg = coutLocal + v;
                float val = 0.0f;
                if (cg < cout) {
                    val = acc[mt][nt][v] + bias[cg];
                    if (relu) val = fmaxf(val, 0.0f);
                }
                o.h[v] = (__bf16)val;
            }
            *(uint4*)&out[(size_t)vox * outStride + coutLocal] = o.q;
        }
    }
}

// ---------------------------------------------------------------------------
// Final fused kernel: 1x1x1 conv (32 -> 6) + 3x3 symmetric eigendecomposition
// + Gram-Schmidt (QR) + SPD reconstruction  Q |Lambda| Q^T.
// ---------------------------------------------------------------------------
__device__ inline float fixv(float v) { return isfinite(v) ? v : 1e-10f; }

__device__ inline void cross3(const float a[3], const float b[3], float c[3]) {
    c[0] = a[1] * b[2] - a[2] * b[1];
    c[1] = a[2] * b[0] - a[0] * b[2];
    c[2] = a[0] * b[1] - a[1] * b[0];
}
__device__ inline float dot3(const float a[3], const float b[3]) {
    return a[0] * b[0] + a[1] * b[1] + a[2] * b[2];
}

__device__ inline void eigvec3(float a00, float a11, float a22,
                               float a01, float a02, float a12,
                               float l, float v[3]) {
    float r0[3] = {a00 - l, a01, a02};
    float r1[3] = {a01, a11 - l, a12};
    float r2[3] = {a02, a12, a22 - l};
    float c0[3], c1[3], c2[3];
    cross3(r0, r1, c0);
    cross3(r1, r2, c1);
    cross3(r0, r2, c2);
    float n0 = dot3(c0, c0), n1 = dot3(c1, c1), n2 = dot3(c2, c2);
    const float* best = c0; float nb = n0;
    if (n1 > nb) { best = c1; nb = n1; }
    if (n2 > nb) { best = c2; nb = n2; }
    float s = rsqrtf(nb + 1e-30f);
    v[0] = best[0] * s; v[1] = best[1] * s; v[2] = best[2] * s;
}

__global__ void final_eig_kernel(const __bf16* __restrict__ act, int stride,
                                 const float* __restrict__ wf,
                                 const float* __restrict__ bf,
                                 float* __restrict__ out) {
    int vox = blockIdx.x * blockDim.x + threadIdx.x;
    if (vox >= NVOX) return;
    const __bf16* h = &act[(size_t)vox * stride];

    float p[6];
#pragma unroll
    for (int j = 0; j < 6; ++j) {
        float s = bf[j];
        for (int c = 0; c < 32; ++c) s += (float)h[c] * wf[j * 32 + c];
        p[j] = s;
    }
    float a00 = fixv(p[0]), a11 = fixv(p[1]), a22 = fixv(p[2]);
    float a01 = fixv(p[3]), a02 = fixv(p[4]), a12 = fixv(p[5]);

    // Closed-form eigenvalues (trigonometric method)
    float q  = (a00 + a11 + a22) * (1.0f / 3.0f);
    float b00 = a00 - q, b11 = a11 - q, b22 = a22 - q;
    float p2 = b00 * b00 + b11 * b11 + b22 * b22 +
               2.0f * (a01 * a01 + a02 * a02 + a12 * a12);
    float pp  = sqrtf(p2 * (1.0f / 6.0f)) + 1e-20f;
    float inv = 1.0f / pp;
    float c00 = b00 * inv, c11 = b11 * inv, c22 = b22 * inv;
    float c01 = a01 * inv, c02 = a02 * inv, c12 = a12 * inv;
    float det = c00 * (c11 * c22 - c12 * c12)
              - c01 * (c01 * c22 - c12 * c02)
              + c02 * (c01 * c12 - c11 * c02);
    float r   = fminf(fmaxf(det * 0.5f, -1.0f), 1.0f);
    float phi = acosf(r) * (1.0f / 3.0f);
    float l0  = q + 2.0f * pp * cosf(phi);
    float l2  = q + 2.0f * pp * cosf(phi + 2.0943951023931953f);
    float l1  = 3.0f * q - l0 - l2;

    // Eigenvectors + modified Gram-Schmidt (mirrors reference qr(eigvecs))
    float q0[3], v1[3], q1[3], q2[3];
    eigvec3(a00, a11, a22, a01, a02, a12, l0, q0);
    eigvec3(a00, a11, a22, a01, a02, a12, l1, v1);
    float d01 = dot3(q0, v1);
    v1[0] -= d01 * q0[0]; v1[1] -= d01 * q0[1]; v1[2] -= d01 * q0[2];
    float s1 = rsqrtf(dot3(v1, v1) + 1e-30f);
    q1[0] = v1[0] * s1; q1[1] = v1[1] * s1; q1[2] = v1[2] * s1;
    cross3(q0, q1, q2);
    float s2 = rsqrtf(dot3(q2, q2) + 1e-30f);
    q2[0] *= s2; q2[1] *= s2; q2[2] *= s2;

    float L[3] = {fabsf(l0), fabsf(l1), fabsf(l2)};
    float Q[3][3] = {{q0[0], q1[0], q2[0]},
                     {q0[1], q1[1], q2[1]},
                     {q0[2], q1[2], q2[2]}};
#pragma unroll
    for (int i = 0; i < 3; ++i)
#pragma unroll
        for (int j = 0; j < 3; ++j) {
            float s = 0.0f;
#pragma unroll
            for (int k = 0; k < 3; ++k) s += L[k] * Q[i][k] * Q[j][k];
            out[(size_t)vox * 9 + i * 3 + j] = s;
        }
}

// ---------------------------------------------------------------------------
// Host launcher
// ---------------------------------------------------------------------------
extern "C" void kernel_launch(void* const* d_in, const int* in_sizes, int n_in,
                              void* d_out, int out_size, void* d_ws, size_t ws_size,
                              hipStream_t stream) {
    (void)in_sizes; (void)n_in; (void)out_size; (void)ws_size;
    static const int CHh[10] = {91, 64, 256, 256, 256, 128, 128, 64, 32, 32};

    const float* x = (const float*)d_in[0];
    char* ws = (char*)d_ws;

    // Workspace layout: [zero page 256B][bufA][bufB][bf16 weights...]
    __bf16* zbuf = (__bf16*)ws;
    const size_t actBytes = (size_t)NVOX * 256 * 2;  // max stride 256, bf16
    __bf16* bufA = (__bf16*)(ws + 256);
    __bf16* bufB = (__bf16*)(ws + 256 + actBytes);
    char*   wptr = ws + 256 + 2 * actBytes;

    // 1) repack + abs input into channels-last bf16 (stride 96), zero halo page
    {
        int n = NVOX * 96;
        repack_input_kernel<<<(n + 255) / 256, 256, 0, stream>>>(x, bufA, zbuf, n);
    }

    __bf16* cur = bufA;
    int curStride = 96;

    // 2) nine conv3d layers via implicit-GEMM WMMA (async LDS staging)
    for (int i = 0; i < 9; ++i) {
        int cin = CHh[i], cout = CHh[i + 1];
        int cinK    = (cin + 31) & ~31;   // K extent (channels zero-padded)
        int coutPad = (cout + 63) & ~63;  // padding to 64-cout tiles

        const float* w = (const float*)d_in[1 + 2 * i];
        const float* b = (const float*)d_in[2 + 2 * i];

        __bf16* wT = (__bf16*)wptr;
        size_t wElems = (size_t)27 * coutPad * cinK;
        wptr += wElems * 2;

        {
            int n = (int)wElems;
            repack_weights_kernel<<<(n + 255) / 256, 256, 0, stream>>>(
                w, wT, cout, cin, coutPad, cinK);
        }

        __bf16* outBuf = (cur == bufA) ? bufB : bufA;
        int outStride = coutPad;

        dim3 grid(NVOX / 256, coutPad / 64);
        conv3d_bf16_wmma_kernel<<<grid, 256, LDS_BYTES, stream>>>(
            cur, curStride, cinK, outBuf, outStride, cout,
            wT, coutPad, b, zbuf, (i < 8) ? 1 : 0);

        cur = outBuf;
        curStride = outStride;
    }

    // 3) 1x1x1 conv + eigendecomposition + SPD reconstruction
    const float* wf = (const float*)d_in[19];
    const float* bf = (const float*)d_in[20];
    final_eig_kernel<<<(NVOX + 255) / 256, 256, 0, stream>>>(
        cur, curStride, wf, bf, (float*)d_out);
}
